// OA_2b_33097017983621
// MI455X (gfx1250) — matrix operand order repaired and checked by reading, not compiled, not run on hardware
//
#include <hip/hip_runtime.h>

// ---------------------------------------------------------------------------
// CDNA5 (gfx1250) fused offset-attention (PCT SA layer) pipeline.
// fp32 inputs; large GEMMs on v_wmma_f32_16x16x32_f16 with f32 accumulation.
// Flash-style two-pass softmax avoids materializing the 537MB attention matrix.
// x_k tiles are staged to LDS with async global->LDS copies (ASYNCcnt).
// ---------------------------------------------------------------------------

typedef __attribute__((ext_vector_type(16))) _Float16 v16h;
typedef __attribute__((ext_vector_type(8)))  _Float16 v8h;
typedef __attribute__((ext_vector_type(8)))  float    v8f;
typedef int v4i_ __attribute__((vector_size(16)));  // matches builtin V4i

constexpr int kB = 8, kCM = 256, kDX = 128, kN = 4096, kCM4 = 64;

#if __has_builtin(__builtin_amdgcn_global_load_async_to_lds_b128) && \
    __has_builtin(__builtin_amdgcn_s_wait_asynccnt)
#define HAVE_ASYNC_LDS 1
#define ASYNC_WAIT(n) __builtin_amdgcn_s_wait_asynccnt(n)
#else
#define HAVE_ASYNC_LDS 0
#define ASYNC_WAIT(n)
#endif

__device__ __forceinline__ v8f wmma_f16(v16h a, v16h b, v8f c) {
  // 8 args: (neg_a, A, neg_b, B, c_mod, C, reuse_a, reuse_b)
  return __builtin_amdgcn_wmma_f32_16x16x32_f16(false, a, false, b, (short)0, c,
                                                false, false);
}

// A fragment: 16 rows x 32 K (f16), row-major source, leading dim ld.
// ISA 7.12.2: lanes 0-15 -> K k0+[0..7] & k0+[16..23]; lanes 16-31 -> +8.
__device__ __forceinline__ v16h frag_A(const _Float16* A, int ld, int r0,
                                       int k0, int lane) {
  int hi = (lane >> 4) & 1;
  const _Float16* p = A + (size_t)(r0 + (lane & 15)) * ld + k0 + hi * 8;
  v8h lo = *(const v8h*)p;
  v8h hh = *(const v8h*)(p + 16);
  return __builtin_shufflevector(lo, hh, 0, 1, 2, 3, 4, 5, 6, 7, 8, 9, 10, 11,
                                 12, 13, 14, 15);
}

// B fragment: 32 K x 16 cols (f16), source stored as Bt[col][k], leading dim ld.
// ISA 7.12.2: lane = col; lanes 0-15 -> K k0+[0..7] & k0+[16..23]; lanes 16-31 -> +8.
__device__ __forceinline__ v16h frag_B(const _Float16* Bt, int ld, int c0,
                                       int k0, int lane) {
  int hi = (lane >> 4) & 1;
  const _Float16* p = Bt + (size_t)(c0 + (lane & 15)) * ld + k0 + hi * 8;
  v8h lo = *(const v8h*)p;
  v8h hh = *(const v8h*)(p + 16);
  return __builtin_shufflevector(lo, hh, 0, 1, 2, 3, 4, 5, 6, 7, 8, 9, 10, 11,
                                 12, 13, 14, 15);
}

// Stage a 64-col x 64-K f16 tile (8 KB, 128B per col) from global to LDS.
// 256 threads x 2 x b128 chunks. Async path uses per-lane VGPR addresses.
__device__ __forceinline__ void stage_tile_8k(const _Float16* __restrict__ src,
                                              _Float16* __restrict__ dst,
                                              int tid) {
#pragma unroll
  for (int j = 0; j < 2; ++j) {
    int chunk = tid * 2 + j;          // 0..511
    int off = chunk * 8;              // halves; 16B granules
#if HAVE_ASYNC_LDS
    __builtin_amdgcn_global_load_async_to_lds_b128(
        (__attribute__((address_space(1))) v4i_*)(src + off),
        (__attribute__((address_space(3))) v4i_*)(dst + off), 0, 0);
#else
    *(v8h*)(dst + off) = *(const v8h*)(src + off);
#endif
  }
}

// ---------------------------------------------------------------------------
// Kernel 1: 1x1-conv projections (small FLOPs). Emits WMMA-layout f16 tensors:
//   xq [b][n][64]  (point-major: A rows for energy)
//   xk [b][n][64]  (point-major: B columns for energy)
//   xv [b][c][n]   (channel-major: A rows for x_r GEMM)
// ---------------------------------------------------------------------------
__global__ __launch_bounds__(256) void proj_kernel(
    const float* __restrict__ q, const float* __restrict__ x,
    const float* __restrict__ Wq, const float* __restrict__ Wk,
    const float* __restrict__ Wv, const float* __restrict__ bv,
    _Float16* __restrict__ xq, _Float16* __restrict__ xk,
    _Float16* __restrict__ xv) {
  __shared__ float tile[128][65];
  int b = blockIdx.x / (kN / 64);
  int p0 = (blockIdx.x % (kN / 64)) * 64;
  int tid = threadIdx.x;
  int p = tid & 63;
  int og = tid >> 6;  // 0..3

  // Stage q tile [128 ch][64 pts]
  for (int i = tid; i < 128 * 64; i += 256) {
    int c = i >> 6, pp = i & 63;
    tile[c][pp] = q[((size_t)b * kDX + c) * kN + p0 + pp];
  }
  __syncthreads();
  // x_q = Wq @ q  (16 outputs per thread)
  for (int o = og * 16; o < og * 16 + 16; ++o) {
    float acc = 0.f;
    for (int c = 0; c < 128; ++c) acc = fmaf(Wq[o * kDX + c], tile[c][p], acc);
    xq[((size_t)b * kN + p0 + p) * kCM4 + o] = (_Float16)acc;
  }
  // x_v = Wv @ q + bv  (64 outputs per thread)
  for (int o = og * 64; o < og * 64 + 64; ++o) {
    float acc = bv[o];
    for (int c = 0; c < 128; ++c) acc = fmaf(Wv[o * kDX + c], tile[c][p], acc);
    xv[((size_t)b * kCM + o) * kN + p0 + p] = (_Float16)acc;
  }
  __syncthreads();
  // x_k = Wk @ x (K=256, two halves through the same LDS tile)
  float acck[16];
#pragma unroll
  for (int i = 0; i < 16; ++i) acck[i] = 0.f;
  for (int half = 0; half < 2; ++half) {
    for (int i = tid; i < 128 * 64; i += 256) {
      int c = i >> 6, pp = i & 63;
      tile[c][pp] = x[((size_t)b * kCM + half * 128 + c) * kN + p0 + pp];
    }
    __syncthreads();
#pragma unroll
    for (int i = 0; i < 16; ++i) {
      int o = og * 16 + i;
      const float* w = Wk + o * kCM + half * 128;
      float acc = acck[i];
      for (int c = 0; c < 128; ++c) acc = fmaf(w[c], tile[c][p], acc);
      acck[i] = acc;
    }
    __syncthreads();
  }
#pragma unroll
  for (int i = 0; i < 16; ++i)
    xk[((size_t)b * kN + p0 + p) * kCM4 + og * 16 + i] = (_Float16)acck[i];
}

__global__ void wt_convert_kernel(const float* __restrict__ Wt,
                                  _Float16* __restrict__ Wt16) {
  int i = blockIdx.x * 256 + threadIdx.x;
  if (i < kCM * kCM) Wt16[i] = (_Float16)Wt[i];
}

// ---------------------------------------------------------------------------
// Kernel 2 (pass A): per-row online softmax statistics (rowmax, rowsumexp).
// Wave owns 16 rows; loops m in tiles of 64. The shared x_k tile is staged to
// LDS with double-buffered async global->LDS copies (all 8 waves reuse it).
// ---------------------------------------------------------------------------
__global__ __launch_bounds__(256) void rowstats_kernel(
    const _Float16* __restrict__ xq, const _Float16* __restrict__ xk,
    float* __restrict__ rowmax, float* __restrict__ rowsum) {
  __shared__ _Float16 kbuf[2][64 * 64];  // two 8 KB tiles: [col][K]
  int b = blockIdx.x / (kN / 128);
  int n0 = (blockIdx.x % (kN / 128)) * 128;
  int tid = threadIdx.x;
  int wave = tid >> 5;
  int lane = tid & 31;
  int r0 = n0 + wave * 16;
  const _Float16* xqb = xq + (size_t)b * kN * kCM4;
  const _Float16* xkb = xk + (size_t)b * kN * kCM4;

  v16h a0 = frag_A(xqb, kCM4, r0, 0, lane);
  v16h a1 = frag_A(xqb, kCM4, r0, 32, lane);

  float rm[8], rs[8];
#pragma unroll
  for (int i = 0; i < 8; ++i) { rm[i] = -3.0e38f; rs[i] = 0.f; }

  stage_tile_8k(xkb, kbuf[0], tid);  // prologue: tile 0

  for (int m0 = 0; m0 < kN; m0 += 64) {
    int cur = (m0 >> 6) & 1;
    bool more = (m0 + 64) < kN;
    if (more) {
      stage_tile_8k(xkb + (size_t)(m0 + 64) * kCM4, kbuf[cur ^ 1], tid);
      ASYNC_WAIT(2);  // current tile done (async ops complete in order)
    } else {
      ASYNC_WAIT(0);
    }
    __syncthreads();
    const _Float16* kb = kbuf[cur];
    v8f e[4];
#pragma unroll
    for (int t = 0; t < 4; ++t) {
      v16h b0 = frag_B(kb, kCM4, t * 16, 0, lane);
      v16h b1 = frag_B(kb, kCM4, t * 16, 32, lane);
      v8f acc = {};
      acc = wmma_f16(a0, b0, acc);
      acc = wmma_f16(a1, b1, acc);
      e[t] = acc;
    }
#pragma unroll
    for (int i = 0; i < 8; ++i) {
      float tm = fmaxf(fmaxf(e[0][i], e[1][i]), fmaxf(e[2][i], e[3][i]));
#pragma unroll
      for (int mask = 1; mask <= 8; mask <<= 1)
        tm = fmaxf(tm, __shfl_xor(tm, mask, 32));
      float newm = fmaxf(rm[i], tm);
      float s = __expf(e[0][i] - newm) + __expf(e[1][i] - newm) +
                __expf(e[2][i] - newm) + __expf(e[3][i] - newm);
#pragma unroll
      for (int mask = 1; mask <= 8; mask <<= 1) s += __shfl_xor(s, mask, 32);
      rs[i] = rs[i] * __expf(rm[i] - newm) + s;
      rm[i] = newm;
    }
    __syncthreads();  // before next stage overwrites the other buffer
  }
  if ((lane & 15) == 0) {
    int rowbase = r0 + (lane >> 4) * 8;
#pragma unroll
    for (int i = 0; i < 8; ++i) {
      rowmax[(size_t)b * kN + rowbase + i] = rm[i];
      rowsum[(size_t)b * kN + rowbase + i] = rs[i];
    }
  }
}

// ---------------------------------------------------------------------------
// Kernel 3 (pass B): recompute E, form P = exp(E-rowmax)/rowsum, accumulate
// colsum[m] and Y[c,m] = sum_n xv[c,n] P[n,m] via WMMA with P staged in LDS.
// Emits u = x - Y/colsum in f16 point-major [b][m][256] for the trans conv.
// Block = (b, 64 columns); wave w: m-sub = w&3, n-half & c-half = w>>2.
// ---------------------------------------------------------------------------
__global__ __launch_bounds__(256) void attn_xr_kernel(
    const _Float16* __restrict__ xq, const _Float16* __restrict__ xk,
    const _Float16* __restrict__ xv, const float* __restrict__ rowmax,
    const float* __restrict__ rowsum, const float* __restrict__ x,
    _Float16* __restrict__ u) {
  __shared__ _Float16 P_lds[4 * 16 * 32];  // [msub][col][K=32]
  __shared__ float colsum_lds[64];
  int b = blockIdx.x / (kN / 64);
  int m0 = (blockIdx.x % (kN / 64)) * 64;
  int w = threadIdx.x >> 5;
  int lane = threadIdx.x & 31;
  int msub = w & 3;
  int whalf = w >> 2;  // n-half for E duty, c-half for GEMM duty
  int hi = lane >> 4;
  if (threadIdx.x < 64) colsum_lds[threadIdx.x] = 0.f;

  const _Float16* xqb = xq + (size_t)b * kN * kCM4;
  const _Float16* xkb = xk + (size_t)b * kN * kCM4;
  const _Float16* xvb = xv + (size_t)b * kCM * kN;
  const float* rmb = rowmax + (size_t)b * kN;
  const float* rsb = rowsum + (size_t)b * kN;

  // Loop-invariant B fragments of the energy GEMM (this wave's 16 columns).
  v16h bE0 = frag_B(xkb, kCM4, m0 + msub * 16, 0, lane);
  v16h bE1 = frag_B(xkb, kCM4, m0 + msub * 16, 32, lane);

  v8f yacc[8];
#pragma unroll
  for (int i = 0; i < 8; ++i) yacc[i] = (v8f){};
  float cs = 0.f;
  _Float16* pbase =
      P_lds + (size_t)(msub * 16 + (lane & 15)) * 32 + whalf * 16 + hi * 8;
  __syncthreads();

  for (int n0 = 0; n0 < kN; n0 += 32) {
    int nblk = n0 + whalf * 16;
    v16h a0 = frag_A(xqb, kCM4, nblk, 0, lane);
    v16h a1 = frag_A(xqb, kCM4, nblk, 32, lane);
    v8f e = {};
    e = wmma_f16(a0, bE0, e);
    e = wmma_f16(a1, bE1, e);
    if (n0 + 32 < kN) {
      __builtin_prefetch(xqb + (size_t)(nblk + 32) * kCM4, 0, 3);
      __builtin_prefetch(xvb + (size_t)(whalf * 128) * kN + n0 + 32, 0, 3);
    }
    v8h pv;
#pragma unroll
    for (int i = 0; i < 8; ++i) {
      int r = nblk + i + hi * 8;  // C-layout row for this lane/vgpr
      float pr = __expf(e[i] - rmb[r]) / rsb[r];
      cs += pr;
      pv[i] = (_Float16)pr;
    }
    *(v8h*)pbase = pv;  // ds_store_b128
    __syncthreads();
    v16h bP = frag_B(P_lds + msub * 16 * 32, 32, 0, 0, lane);  // ds_load
#pragma unroll
    for (int i = 0; i < 8; ++i) {
      v16h aV = frag_A(xvb, kN, whalf * 128 + i * 16, n0, lane);
      yacc[i] = wmma_f16(aV, bP, yacc[i]);
    }
    __syncthreads();
  }

  // Column sums: combine lane halves (rows split), then both n-half waves.
  cs += __shfl_xor(cs, 16, 32);
  if (hi == 0) atomicAdd(&colsum_lds[msub * 16 + (lane & 15)], cs);
  __syncthreads();

  int col = m0 + msub * 16 + (lane & 15);
  float inv = 1.0f / (1e-9f + colsum_lds[msub * 16 + (lane & 15)]);
#pragma unroll
  for (int i = 0; i < 8; ++i) {
    int c0 = whalf * 128 + i * 16 + hi * 8;
    v8h uv;
#pragma unroll
    for (int j = 0; j < 8; ++j) {
      float xr = yacc[i][j] * inv;
      float xval = x[((size_t)b * kCM + c0 + j) * kN + col];
      uv[j] = (_Float16)(xval - xr);
    }
    *(v8h*)(u + ((size_t)b * kN + col) * kCM + c0) = uv;
  }
}

// ---------------------------------------------------------------------------
// Kernel 4: t = Wt @ u + bt (WMMA, K=256), fused per-channel sum/sumsq for BN.
// ---------------------------------------------------------------------------
__global__ __launch_bounds__(256) void tgemm_kernel(
    const _Float16* __restrict__ Wt16, const _Float16* __restrict__ u,
    const float* __restrict__ bt, float* __restrict__ tpm,
    float* __restrict__ chsum, float* __restrict__ chsumsq) {
  int b = blockIdx.x / (kN / 64);
  int m0 = (blockIdx.x % (kN / 64)) * 64;
  int w = threadIdx.x >> 5;
  int lane = threadIdx.x & 31;
  int msub = w & 3;
  int chalf = w >> 2;
  int hi = lane >> 4;
  const _Float16* ub = u + ((size_t)b * kN + m0 + msub * 16) * kCM;

  v8f acc[8];
#pragma unroll
  for (int i = 0; i < 8; ++i) acc[i] = (v8f){};
#pragma unroll 2
  for (int k0 = 0; k0 < kCM; k0 += 32) {
    v16h bfr = frag_B(ub, kCM, 0, k0, lane);
#pragma unroll
    for (int i = 0; i < 8; ++i) {
      v16h afr = frag_A(Wt16, kCM, chalf * 128 + i * 16, k0, lane);
      acc[i] = wmma_f16(afr, bfr, acc[i]);
    }
  }
  int col = m0 + msub * 16 + (lane & 15);
#pragma unroll
  for (int i = 0; i < 8; ++i) {
    int c0 = chalf * 128 + i * 16 + hi * 8;
    float vals[8];
#pragma unroll
    for (int j = 0; j < 8; ++j) vals[j] = acc[i][j] + bt[c0 + j];
    float* dst = tpm + ((size_t)b * kN + col) * kCM + c0;
#pragma unroll
    for (int j = 0; j < 8; ++j) dst[j] = vals[j];
#pragma unroll
    for (int j = 0; j < 8; ++j) {
      float s = vals[j], ss = vals[j] * vals[j];
#pragma unroll
      for (int mask = 1; mask <= 8; mask <<= 1) {
        s += __shfl_xor(s, mask, 32);
        ss += __shfl_xor(ss, mask, 32);
      }
      if ((lane & 15) == 0) {
        atomicAdd(&chsum[c0 + j], s);
        atomicAdd(&chsumsq[c0 + j], ss);
      }
    }
  }
}

// ---------------------------------------------------------------------------
// Kernel 5: BN scale/shift from batch statistics (biased variance).
// ---------------------------------------------------------------------------
__global__ void bn_finalize_kernel(const float* __restrict__ chsum,
                                   const float* __restrict__ chsumsq,
                                   const float* __restrict__ gamma,
                                   const float* __restrict__ beta,
                                   float* __restrict__ scale,
                                   float* __restrict__ shift) {
  int c = threadIdx.x;
  float inv_n = 1.0f / (float)(kB * kN);
  float mean = chsum[c] * inv_n;
  float var = chsumsq[c] * inv_n - mean * mean;
  float sc = gamma[c] * rsqrtf(var + 1e-5f);
  scale[c] = sc;
  shift[c] = beta[c] - mean * sc;
}

// ---------------------------------------------------------------------------
// Kernel 6: BN + ReLU, LDS-transposed writeback [b][m][c] -> out [b][c][n].
// ---------------------------------------------------------------------------
__global__ __launch_bounds__(256) void bn_apply_kernel(
    const float* __restrict__ tpm, const float* __restrict__ scale,
    const float* __restrict__ shift, float* __restrict__ out) {
  __shared__ float tile[64][65];
  int tilesPerB = (kN / 64) * (kCM / 64);
  int b = blockIdx.x / tilesPerB;
  int rem = blockIdx.x % tilesPerB;
  int n0 = (rem / (kCM / 64)) * 64;
  int c0 = (rem % (kCM / 64)) * 64;
  int tid = threadIdx.x;
#pragma unroll
  for (int i = 0; i < 16; ++i) {
    int e = i * 256 + tid;
    int p = e >> 6, c = e & 63;
    tile[p][c] = tpm[((size_t)b * kN + n0 + p) * kCM + c0 + c];
  }
  __syncthreads();
#pragma unroll
  for (int i = 0; i < 16; ++i) {
    int e = i * 256 + tid;
    int c = e >> 6, p = e & 63;
    float v = tile[p][c] * scale[c0 + c] + shift[c0 + c];
    out[((size_t)b * kCM + c0 + c) * kN + n0 + p] = fmaxf(v, 0.f);
  }
}

// ---------------------------------------------------------------------------
extern "C" void kernel_launch(void* const* d_in, const int* in_sizes, int n_in,
                              void* d_out, int out_size, void* d_ws,
                              size_t ws_size, hipStream_t stream) {
  (void)in_sizes; (void)n_in; (void)out_size; (void)ws_size;
  const float* q     = (const float*)d_in[0];
  const float* x     = (const float*)d_in[1];
  const float* Wq    = (const float*)d_in[2];
  const float* Wk    = (const float*)d_in[3];
  const float* Wv    = (const float*)d_in[4];
  const float* bv    = (const float*)d_in[5];
  const float* Wt    = (const float*)d_in[6];
  const float* bt    = (const float*)d_in[7];
  const float* gamma = (const float*)d_in[8];
  const float* beta  = (const float*)d_in[9];
  float* out = (float*)d_out;

  char* w = (char*)d_ws;
  _Float16* xq16 = (_Float16*)w; w += (size_t)kB * kN * kCM4 * 2;   // 4 MB
  _Float16* xk16 = (_Float16*)w; w += (size_t)kB * kN * kCM4 * 2;   // 4 MB
  _Float16* xv16 = (_Float16*)w; w += (size_t)kB * kCM * kN * 2;    // 16 MB
  _Float16* wt16 = (_Float16*)w; w += (size_t)kCM * kCM * 2;        // 128 KB
  float* rowmax  = (float*)w;    w += (size_t)kB * kN * 4;          // 128 KB
  float* rowsum  = (float*)w;    w += (size_t)kB * kN * 4;          // 128 KB
  _Float16* u16  = (_Float16*)w; w += (size_t)kB * kN * kCM * 2;    // 16 MB
  float* tpm     = (float*)w;    w += (size_t)kB * kN * kCM * 4;    // 32 MB
  float* chsum   = (float*)w;    w += kCM * 4;
  float* chsumsq = (float*)w;    w += kCM * 4;
  float* scale   = (float*)w;    w += kCM * 4;
  float* shift   = (float*)w;    w += kCM * 4;

  (void)hipMemsetAsync(chsum, 0, 2 * kCM * sizeof(float), stream);  // BN sums

  proj_kernel<<<kB * (kN / 64), 256, 0, stream>>>(q, x, Wq, Wk, Wv, bv, xq16,
                                                  xk16, xv16);
  wt_convert_kernel<<<(kCM * kCM + 255) / 256, 256, 0, stream>>>(Wt, wt16);
  rowstats_kernel<<<kB * (kN / 128), 256, 0, stream>>>(xq16, xk16, rowmax,
                                                       rowsum);
  attn_xr_kernel<<<kB * (kN / 64), 256, 0, stream>>>(xq16, xk16, xv16, rowmax,
                                                     rowsum, x, u16);
  tgemm_kernel<<<kB * (kN / 64), 256, 0, stream>>>(wt16, u16, bt, tpm, chsum,
                                                   chsumsq);
  bn_finalize_kernel<<<1, kCM, 0, stream>>>(chsum, chsumsq, gamma, beta, scale,
                                            shift);
  bn_apply_kernel<<<kB * (kN / 64) * (kCM / 64), 256, 0, stream>>>(tpm, scale,
                                                                   shift, out);
}
